// GroupVectorQuantizer_42271068127277
// MI455X (gfx1250) — matrix operand order repaired and checked by reading, not compiled, not run on hardware
//
#include <hip/hip_runtime.h>

// Problem constants (from reference)
#define G_     8
#define K_     1024
#define D_     64
#define B_     16
#define T_     4000
#define CF_    512                    // C*F = G_*D_
#define NTOT_  (B_*CF_*T_)            // 32768000 elements per output tensor
#define TOKPB  256                    // tokens per block
#define NTHR   128                    // 4 waves

typedef __attribute__((ext_vector_type(16))) __bf16 v16bf;
typedef __attribute__((ext_vector_type(8)))  float  v8f;

// ---------------------------------------------------------------------------
// Prep 1: pack codebook into the WMMA B-operand register image (bf16).
// Layout: bimg[g][kt][ko][lane][j] (dwords), kt = 16-code tile (64 tiles),
// ko = K-slab (D=64 -> two 32-deep slabs), lane 0..31, j = VGPR 0..7.
// B element (k,n) of a 32x16 tile: lane = (k>>4)*16 + n, VGPR j = (k&15)>>1,
// halfword = k&1.  => dword j of lane l holds k = (l>>4)*16 + 2j, 2j+1 for
// code n = l&15.
// ---------------------------------------------------------------------------
__global__ void vq_prep_bimg(const float* __restrict__ cb,
                             unsigned* __restrict__ bimg) {
  unsigned tid = blockIdx.x * blockDim.x + threadIdx.x;  // 2^17 threads
  if (tid >= G_ * 64u * 32u * 8u) return;
  unsigned j    = tid & 7u;
  unsigned lane = (tid >> 3) & 31u;
  unsigned kt   = (tid >> 8) & 63u;
  unsigned g    = tid >> 14;
  unsigned code = kt * 16u + (lane & 15u);
  unsigned kb   = ((lane >> 4) << 4) + 2u * j;           // k within 32-slab
  const float* row = cb + (size_t)(g * K_ + code) * D_;
#pragma unroll
  for (unsigned ko = 0; ko < 2; ++ko) {
    unsigned dd = ko * 32u + kb;
    union { __bf16 h[2]; unsigned u; } pk;
    pk.h[0] = (__bf16)row[dd];
    pk.h[1] = (__bf16)row[dd + 1];
    bimg[((g * 64u + kt) * 2u + ko) * 256u + lane * 8u + j] = pk.u;
  }
}

// Prep 2: e2h[g][k] = 0.5 * ||E_gk||^2  (argmin(e2 - 2 x.e) == argmin(.5e2 - x.e))
__global__ void vq_prep_e2(const float* __restrict__ cb,
                           float* __restrict__ e2h) {
  unsigned tid = blockIdx.x * blockDim.x + threadIdx.x;
  if (tid >= G_ * K_) return;
  const float* row = cb + (size_t)tid * D_;
  float s = 0.f;
#pragma unroll 8
  for (int d = 0; d < D_; ++d) { float v = row[d]; s += v * v; }
  e2h[tid] = 0.5f * s;
}

// One 16x16x64 accumulation step against code tile KT, then running-min update.
#define VQ_STEP(KT, BA, BB, E2V)                                              \
  {                                                                           \
    _Pragma("unroll")                                                         \
    for (int s = 0; s < 4; ++s) {                                             \
      v8f acc = {0.f, 0.f, 0.f, 0.f, 0.f, 0.f, 0.f, 0.f};                     \
      acc = __builtin_amdgcn_wmma_f32_16x16x32_bf16(                          \
          false, A[s][0], false, (BA), (short)0, acc, false, false);          \
      acc = __builtin_amdgcn_wmma_f32_16x16x32_bf16(                          \
          false, A[s][1], false, (BB), (short)0, acc, false, false);          \
      unsigned code = (KT) * 16u + m;                                         \
      _Pragma("unroll")                                                       \
      for (int r = 0; r < 8; ++r) {                                           \
        float sc  = (E2V) - acc[r];                                           \
        bool take = sc < minv[s][r];                                          \
        minv[s][r] = take ? sc : minv[s][r];                                  \
        mini[s][r] = take ? code : mini[s][r];                                \
      }                                                                       \
    }                                                                         \
  }

// ---------------------------------------------------------------------------
// Main: one block = 256 consecutive tokens t of one (g, b). 4 waves, each
// owns 4 A-tiles (16 tokens x 64 dims bf16 in registers). Sweep 64 code
// tiles with double-buffered B in VGPRs; 8 v_wmma per tile per wave.
// ---------------------------------------------------------------------------
__launch_bounds__(NTHR, 1)
__global__ void vq_main(const float* __restrict__ x,
                        const float* __restrict__ cb,
                        const unsigned* __restrict__ bimg,
                        const float* __restrict__ e2h,
                        float* __restrict__ out) {
  __shared__ int   idx_lds[TOKPB];
  __shared__ float staged[128 * 68];           // 128 tokens x 64 f32, padded

  const unsigned bx = blockIdx.x;
  const unsigned g  = bx >> 8;                 // 256 blocks per group
  const unsigned b  = (bx >> 4) & 15u;
  const unsigned bt = bx & 15u;
  const unsigned t0 = bt * TOKPB;
  const unsigned validTok = (t0 + TOKPB <= T_) ? TOKPB : (T_ - t0);  // 256/160
  const unsigned tid  = threadIdx.x;
  const unsigned w    = tid >> 5;
  const unsigned lane = tid & 31u;
  const unsigned m    = lane & 15u;            // M row / N col index
  const unsigned h    = lane >> 4;             // lane-half

  // ---- build A operands (contiguous k = 16h + e, same k order as bimg), zero for invalid tiles
  v16bf A[4][2];
#pragma unroll
  for (int s = 0; s < 4; ++s)
#pragma unroll
    for (int ko = 0; ko < 2; ++ko)
#pragma unroll
      for (int e = 0; e < 16; ++e) A[s][ko][e] = (__bf16)0.0f;

#pragma unroll
  for (int s = 0; s < 4; ++s) {
    unsigned it = (unsigned)s * 4u + w;        // tile of 16 tokens in block
    if (it * 16u < validTok) {
      unsigned t = t0 + it * 16u + m;
      const float* xb = x + (size_t)(b * CF_ + g * D_) * T_ + t;
#pragma unroll
      for (int j = 0; j < 8; ++j) {
        unsigned kb = h * 16u + 2u * (unsigned)j;
#pragma unroll
        for (int ko = 0; ko < 2; ++ko) {
          unsigned dd = ko * 32u + kb;
          float f0 = __builtin_nontemporal_load(xb + (size_t)dd * T_);
          float f1 = __builtin_nontemporal_load(xb + (size_t)(dd + 1) * T_);
          A[s][ko][2 * j]     = (__bf16)f0;
          A[s][ko][2 * j + 1] = (__bf16)f1;
        }
      }
    }
  }

  float    minv[4][8];
  unsigned mini[4][8];
#pragma unroll
  for (int s = 0; s < 4; ++s)
#pragma unroll
    for (int r = 0; r < 8; ++r) { minv[s][r] = __builtin_inff(); mini[s][r] = 0u; }

  const unsigned* bp  = bimg + (size_t)g * 64u * 512u + lane * 8u;
  const float*    e2p = e2h + g * K_ + m;

  v16bf B0a = *(const v16bf*)bp;
  v16bf B0b = *(const v16bf*)(bp + 256);
  float e2a = e2p[0];

#pragma unroll 1
  for (unsigned kt = 0; kt < 64u; kt += 2u) {
    const unsigned* bp1 = bp + (kt + 1u) * 512u;
    v16bf B1a = *(const v16bf*)bp1;
    v16bf B1b = *(const v16bf*)(bp1 + 256);
    float e2b = e2p[(kt + 1u) * 16u];

    VQ_STEP(kt, B0a, B0b, e2a);

    if (kt + 2u < 64u) {
      const unsigned* bp2 = bp + (kt + 2u) * 512u;
      B0a = *(const v16bf*)bp2;
      B0b = *(const v16bf*)(bp2 + 256);
      e2a = e2p[(kt + 2u) * 16u];
    }

    VQ_STEP(kt + 1u, B1a, B1b, e2b);
  }

  // ---- cross-lane argmin within each 16-lane N-group (wave32 shuffles)
#pragma unroll
  for (int s = 0; s < 4; ++s) {
    unsigned it = (unsigned)s * 4u + w;
    bool tv = (it * 16u < validTok);
#pragma unroll
    for (int r = 0; r < 8; ++r) {
      float    v = minv[s][r];
      unsigned i = mini[s][r];
#pragma unroll
      for (int msk = 1; msk < 16; msk <<= 1) {
        float    v2 = __shfl_xor(v, msk, 32);
        unsigned i2 = (unsigned)__shfl_xor((int)i, msk, 32);
        if (v2 < v || (v2 == v && i2 < i)) { v = v2; i = i2; }
      }
      if (tv && m == (unsigned)r)
        idx_lds[it * 16u + (unsigned)r + 8u * h] = (int)i;   // token M=r+8h
    }
  }
  __syncthreads();

  // ---- gather winning code vectors (coalesced) and stream both outputs
  const size_t outBase = (size_t)(b * CF_ + g * D_) * T_;
#pragma unroll 1
  for (int c = 0; c < 2; ++c) {
    unsigned tokBase = (unsigned)c * 128u;
#pragma unroll 1
    for (int i = 0; i < 16; ++i) {             // 8 tokens x 16 lanes each
      unsigned row = (unsigned)i * 8u + (tid >> 4);
      unsigned tl  = tokBase + row;
      unsigned dd0 = (tid & 15u) * 4u;
      if (tl < validTok) {
        int code = idx_lds[tl];
        const float4 q =
            *(const float4*)(cb + (size_t)(g * K_ + (unsigned)code) * D_ + dd0);
        *(float4*)(&staged[row * 68u + dd0]) = q;
      }
    }
    __syncthreads();
    unsigned tl = tokBase + tid;
    if (tl < validTok) {
      unsigned t = t0 + tl;
      float* o0 = out + outBase + t;
      float* o1 = o0 + (size_t)NTOT_;
#pragma unroll
      for (int dd = 0; dd < 64; ++dd) {
        float v = staged[tid * 68u + (unsigned)dd];
        __builtin_nontemporal_store(v, o0 + (size_t)dd * T_);
        __builtin_nontemporal_store(v, o1 + (size_t)dd * T_);
      }
    }
    __syncthreads();
  }
}

// ---------------------------------------------------------------------------
extern "C" void kernel_launch(void* const* d_in, const int* in_sizes, int n_in,
                              void* d_out, int out_size, void* d_ws,
                              size_t ws_size, hipStream_t stream) {
  (void)in_sizes; (void)n_in; (void)out_size; (void)ws_size;
  const float* x  = (const float*)d_in[0];   // [B, C, F, T] f32
  const float* cb = (const float*)d_in[1];   // [G, K, D]    f32

  unsigned* bimg = (unsigned*)d_ws;                         // 1 MiB
  float*    e2h  = (float*)((char*)d_ws + (1u << 20));      // 32 KiB

  vq_prep_bimg<<<512, 256, 0, stream>>>(cb, bimg);
  vq_prep_e2<<<32, 256, 0, stream>>>(cb, e2h);
  vq_main<<<G_ * B_ * 16, NTHR, 0, stream>>>(x, cb, bimg, e2h, (float*)d_out);
}
